// APPNPModel_82678120448256
// MI455X (gfx1250) — matrix operand order repaired and checked by reading, not compiled, not run on hardware
//
#include <hip/hip_runtime.h>

typedef __attribute__((ext_vector_type(16))) __bf16 v16bf;
typedef __attribute__((ext_vector_type(8)))  __bf16 v8bf;
typedef __attribute__((ext_vector_type(8)))  float  v8f;

#define NNODES 100000
#define NEDGES 3200000
#define FIN    512
#define FHID   256
#define FOUT   64
#define ALPHA_F 0.1f

// ---------------------------------------------------------------------------
// Pack a row-major [K x Nn] fp32 weight matrix into the CDNA5 WMMA B-operand
// tile layout (bf16). For a 32x16 (KxN) tile: lane L holds column n = L%16,
// elements i=0..15 hold K = tile_k*32 + (i>=8 ? 16 : 0) + (L>=16 ? 8 : 0) + (i&7).
// Tile order: tileIdx = nt * (K/32) + kt ; each tile is 32 lanes * 16 elems,
// so for fixed nt the kt tiles are contiguous with a 512-element stride.
// ---------------------------------------------------------------------------
__global__ void pack_b_tiles(const float* __restrict__ W, __bf16* __restrict__ out,
                             int K, int Nn) {
  int o = blockIdx.x * blockDim.x + threadIdx.x;
  int total = K * Nn;
  if (o >= total) return;
  int ktiles = K >> 5;
  int i    = o & 15;
  int lane = (o >> 4) & 31;
  int tile = o >> 9;                 // 512 elements per tile
  int kt = tile % ktiles;
  int nt = tile / ktiles;
  int n = nt * 16 + (lane & 15);
  int k = kt * 32 + ((i >> 3) << 4) + ((lane >> 4) << 3) + (i & 7);
  out[o] = (__bf16)W[(size_t)k * Nn + n];
}

// ---------------------------------------------------------------------------
// Fused MLP head: h0 = relu(x @ W1 + b1) @ W2 + b2    (per 16-row tile)
// Block = 128 threads = 4 waves. GEMM1: kt-outer loop loads each A tile from
// LDS once and feeds 4 accumulators (4 HID tiles per wave) -> WMMA-dense
// inner loop with constant-offset B loads. GEMM2: 1 OUT tile per wave.
// ---------------------------------------------------------------------------
__global__ void __launch_bounds__(128)
mlp_head(const float* __restrict__ x, const __bf16* __restrict__ W1p,
         const float* __restrict__ b1, const __bf16* __restrict__ W2p,
         const float* __restrict__ b2, float* __restrict__ h0) {
  __shared__ __align__(16) __bf16 Abuf[16 * FIN];   // 16 KB: 16 rows of x (bf16)
  __shared__ __align__(16) __bf16 Hbuf[16 * FHID];  //  8 KB: relu intermediate

  const int tid = threadIdx.x;
  const int m0  = blockIdx.x * 16;

  // stage 16 rows of x into LDS as bf16 (coalesced)
  for (int idx = tid; idx < 16 * FIN; idx += 128) {
    int row = idx >> 9, col = idx & 511;
    Abuf[idx] = (__bf16)x[(size_t)(m0 + row) * FIN + col];
  }
  __syncthreads();

  const int wave = tid >> 5, lane = tid & 31;
  const int m    = lane & 15;          // A row / C column index
  const int koff = (lane >> 4) << 3;   // 0 (lanes 0-15) or 8 (lanes 16-31)

  // ---- GEMM1: x(16x512) @ W1(512x256), bias + relu -> Hbuf ----
  {
    // base pointers: tile (nt, kt) starts at ((nt*16 + kt)*32 + lane)*16
    const __bf16* b0 = W1p + ((size_t)(wave * 4 + 0) * (FIN / 32) * 32 + lane) * 16;
    const __bf16* b1p = W1p + ((size_t)(wave * 4 + 1) * (FIN / 32) * 32 + lane) * 16;
    const __bf16* b2p = W1p + ((size_t)(wave * 4 + 2) * (FIN / 32) * 32 + lane) * 16;
    const __bf16* b3p = W1p + ((size_t)(wave * 4 + 3) * (FIN / 32) * 32 + lane) * 16;
    const __bf16* arow = &Abuf[m * FIN + koff];

    v8f acc0 = {}, acc1 = {}, acc2 = {}, acc3 = {};
    #pragma unroll
    for (int kt = 0; kt < FIN / 32; ++kt) {
      v8bf alo = *(const v8bf*)(arow + kt * 32);        // K = kb .. kb+7
      v8bf ahi = *(const v8bf*)(arow + kt * 32 + 16);   // K = kb+16 .. kb+23
      v16bf a;
      #pragma unroll
      for (int i = 0; i < 8; ++i) { a[i] = alo[i]; a[i + 8] = ahi[i]; }
      v16bf bb0 = *(const v16bf*)(b0  + (size_t)kt * 512);
      v16bf bb1 = *(const v16bf*)(b1p + (size_t)kt * 512);
      v16bf bb2 = *(const v16bf*)(b2p + (size_t)kt * 512);
      v16bf bb3 = *(const v16bf*)(b3p + (size_t)kt * 512);
      acc0 = __builtin_amdgcn_wmma_f32_16x16x32_bf16(false, a, false, bb0, (short)0, acc0, false, false);
      acc1 = __builtin_amdgcn_wmma_f32_16x16x32_bf16(false, a, false, bb1, (short)0, acc1, false, false);
      acc2 = __builtin_amdgcn_wmma_f32_16x16x32_bf16(false, a, false, bb2, (short)0, acc2, false, false);
      acc3 = __builtin_amdgcn_wmma_f32_16x16x32_bf16(false, a, false, bb3, (short)0, acc3, false, false);
    }

    // bias + relu -> Hbuf (C layout: lane column m, rows r + koff)
    v8f accs[4] = {acc0, acc1, acc2, acc3};
    #pragma unroll
    for (int t = 0; t < 4; ++t) {
      const int ncol = (wave * 4 + t) * 16 + m;
      const float bias = b1[ncol];
      #pragma unroll
      for (int r = 0; r < 8; ++r) {
        float v = accs[t][r] + bias;
        v = v > 0.f ? v : 0.f;
        Hbuf[(r + koff) * FHID + ncol] = (__bf16)v;
      }
    }
  }
  __syncthreads();

  // ---- GEMM2: hid(16x256) @ W2(256x64), bias -> h0 ----
  {
    const __bf16* bbase = W2p + ((size_t)wave * (FHID / 32) * 32 + lane) * 16;
    const __bf16* arow  = &Hbuf[m * FHID + koff];
    v8f acc = {};
    #pragma unroll
    for (int kt = 0; kt < FHID / 32; ++kt) {
      v8bf alo = *(const v8bf*)(arow + kt * 32);
      v8bf ahi = *(const v8bf*)(arow + kt * 32 + 16);
      v16bf a;
      #pragma unroll
      for (int i = 0; i < 8; ++i) { a[i] = alo[i]; a[i + 8] = ahi[i]; }
      v16bf b = *(const v16bf*)(bbase + (size_t)kt * 512);
      acc = __builtin_amdgcn_wmma_f32_16x16x32_bf16(false, a, false, b, (short)0, acc, false, false);
    }
    const int ncol = wave * 16 + m;
    const float bias = b2[ncol];
    #pragma unroll
    for (int r = 0; r < 8; ++r) {
      h0[(size_t)(m0 + r + koff) * FOUT + ncol] = acc[r] + bias;
    }
  }
}

// ---------------------------------------------------------------------------
// Degree / norm kernels. deg starts at 1.0 (self-loop), atomically counts
// incoming edges, then converted in-place to D^-1/2.
// ---------------------------------------------------------------------------
__global__ void deg_init(float* __restrict__ deg) {
  int i = blockIdx.x * blockDim.x + threadIdx.x;
  if (i < NNODES) deg[i] = 1.0f;                    // self-loop
}

__global__ void deg_count(const int* __restrict__ dst, float* __restrict__ deg) {
  int e = blockIdx.x * blockDim.x + threadIdx.x;
  if (e < NEDGES) atomicAdd(&deg[dst[e]], 1.0f);
}

__global__ void deg_to_dinv(float* __restrict__ deg) {
  int i = blockIdx.x * blockDim.x + threadIdx.x;
  if (i < NNODES) {
    float d = deg[i];
    deg[i] = d > 0.f ? rsqrtf(d) : 0.f;
  }
}

__global__ void zero_buf(float* __restrict__ p, int n) {
  int i = blockIdx.x * blockDim.x + threadIdx.x;
  if (i < n) p[i] = 0.f;
}

// ---------------------------------------------------------------------------
// Edge scatter: 32 lanes per edge, 2 features per lane (float2 gather =
// 256 contiguous bytes per source row; src/dst/dinv loads are wave-uniform).
// agg (25.6 MB) and h (25.6 MB) are L2-resident on MI455X (192 MB L2).
// ---------------------------------------------------------------------------
__global__ void edge_scatter(const int* __restrict__ src, const int* __restrict__ dst,
                             const float* __restrict__ dinv, const float* __restrict__ h,
                             float* __restrict__ agg) {
  long long gid = (long long)blockIdx.x * blockDim.x + threadIdx.x;
  int e = (int)(gid >> 5);
  if (e >= NEDGES) return;
  int c = ((int)gid & 31) * 2;
  int s = src[e], d = dst[e];
  float nrm = dinv[s] * dinv[d];
  const float2 hv = *(const float2*)&h[(size_t)s * FOUT + c];
  atomicAdd(&agg[(size_t)d * FOUT + c],     hv.x * nrm);
  atomicAdd(&agg[(size_t)d * FOUT + c + 1], hv.y * nrm);
}

// Fused: self-loop term + teleport + reset agg for the next iteration.
__global__ void combine_step(const float* __restrict__ agg, const float* __restrict__ hcur,
                             const float* __restrict__ h0, const float* __restrict__ dinv,
                             float* __restrict__ hnext, float* __restrict__ aggz) {
  int i = blockIdx.x * blockDim.x + threadIdx.x;
  if (i >= NNODES * FOUT) return;
  int node = i >> 6;
  float di = dinv[node];
  float v = (1.0f - ALPHA_F) * (agg[i] + hcur[i] * di * di) + ALPHA_F * h0[i];
  hnext[i] = v;
  aggz[i] = 0.f;
}

// ---------------------------------------------------------------------------
extern "C" void kernel_launch(void* const* d_in, const int* in_sizes, int n_in,
                              void* d_out, int out_size, void* d_ws, size_t ws_size,
                              hipStream_t stream) {
  (void)in_sizes; (void)n_in; (void)out_size; (void)ws_size;
  const float* x  = (const float*)d_in[0];
  const int*   ei = (const int*)d_in[1];         // edge_index [2][E]
  const float* W1 = (const float*)d_in[2];
  const float* b1 = (const float*)d_in[3];
  const float* W2 = (const float*)d_in[4];
  const float* b2 = (const float*)d_in[5];
  const int* src = ei;
  const int* dst = ei + NEDGES;

  char* ws = (char*)d_ws;
  size_t off = 0;
  auto carve = [&](size_t bytes) -> char* {
    char* p = ws + off;
    off = (off + bytes + 255) & ~(size_t)255;
    return p;
  };
  __bf16* W1p = (__bf16*)carve((size_t)FIN * FHID * 2);      // 256 KB
  __bf16* W2p = (__bf16*)carve((size_t)FHID * FOUT * 2);     //  32 KB
  float*  dinv = (float*)carve((size_t)NNODES * 4);          // 400 KB
  float*  h0   = (float*)carve((size_t)NNODES * FOUT * 4);   // 25.6 MB
  float*  hA   = (float*)carve((size_t)NNODES * FOUT * 4);
  float*  hB   = (float*)carve((size_t)NNODES * FOUT * 4);
  float*  agg  = (float*)carve((size_t)NNODES * FOUT * 4);

  // 1) pack weights to bf16 WMMA B-tile layout
  pack_b_tiles<<<(FIN * FHID + 255) / 256, 256, 0, stream>>>(W1, W1p, FIN, FHID);
  pack_b_tiles<<<(FHID * FOUT + 255) / 256, 256, 0, stream>>>(W2, W2p, FHID, FOUT);

  // 2) fused MLP head via WMMA (6250 row tiles of 16)
  mlp_head<<<NNODES / 16, 128, 0, stream>>>(x, W1p, b1, W2p, b2, h0);

  // 3) gcn_norm: degrees (with self-loop) -> D^-1/2
  deg_init<<<(NNODES + 255) / 256, 256, 0, stream>>>(dinv);
  deg_count<<<(NEDGES + 255) / 256, 256, 0, stream>>>(dst, dinv);
  deg_to_dinv<<<(NNODES + 255) / 256, 256, 0, stream>>>(dinv);

  // 4) K propagation steps (agg zeroed once, then recycled by combine_step)
  zero_buf<<<(NNODES * FOUT + 255) / 256, 256, 0, stream>>>(agg, NNODES * FOUT);

  const float* cur = h0;
  float* bufs[2] = {hA, hB};
  const long long scat_threads = (long long)NEDGES * 32;
  const unsigned scat_blocks = (unsigned)((scat_threads + 255) / 256);
  for (int kstep = 0; kstep < 10; ++kstep) {
    float* next = (kstep == 9) ? (float*)d_out : bufs[kstep & 1];
    edge_scatter<<<scat_blocks, 256, 0, stream>>>(src, dst, dinv, cur, agg);
    combine_step<<<(NNODES * FOUT + 255) / 256, 256, 0, stream>>>(agg, cur, h0, dinv, next, agg);
    cur = next;
  }
}
